// Stacked_Decoder_7876970021436
// MI455X (gfx1250) — matrix operand order repaired and checked by reading, not compiled, not run on hardware
//
#include <hip/hip_runtime.h>
#include <cstdint>
#include <cstddef>

// ---------------- problem constants (match reference) ----------------
#define BB   8
#define TT   1024
#define EE   1024
#define NHH  16
#define DHH  64
#define FFF  4096
#define LLL  6
#define MR   (BB * TT)      // 8192 rows of activations

typedef __attribute__((ext_vector_type(16))) __bf16 v16bf;
typedef __attribute__((ext_vector_type(8)))  float  v8f;
typedef __attribute__((ext_vector_type(4)))  unsigned u32x4;
typedef __attribute__((ext_vector_type(4)))  int      i32x4;
typedef __attribute__((ext_vector_type(8)))  int      i32x8;

union Frag16 { v16bf v; unsigned u[8]; };

// Tensor Data Mover availability (device pass only); 5-arg on clang-22 (ROCm 7.2),
// 6-arg on clang>=23 (amdgpu-toolchain therock headers).
#if defined(__HIP_DEVICE_COMPILE__) && __has_builtin(__builtin_amdgcn_tensor_load_to_lds) && \
    __has_builtin(__builtin_amdgcn_s_wait_tensorcnt)
#define ATHENA_TDM 1
#else
#define ATHENA_TDM 0
#endif

// --- compile-probe diagnostics (show up in stderr, harmless) ---
#define ATH_STR2(x) #x
#define ATH_STR(x) ATH_STR2(x)
#if defined(__HIP_DEVICE_COMPILE__)
#if ATHENA_TDM
#pragma message("ATHENA probe: TDM path ENABLED (tensor_load_to_lds), clang " ATH_STR(__clang_major__))
#else
#pragma message("ATHENA probe: TDM path DISABLED (fallback copy), clang " ATH_STR(__clang_major__))
#endif
#endif

__device__ __forceinline__ unsigned pack2bf(float a, float b) {
    __bf16 x = (__bf16)a, y = (__bf16)b;
    unsigned short ux = __builtin_bit_cast(unsigned short, x);
    unsigned short uy = __builtin_bit_cast(unsigned short, y);
    return (unsigned)ux | ((unsigned)uy << 16);
}

// ---------------- LayerNorm: one block (256 thr) per row of E=1024 ----------------
__global__ void ln_kernel(const float* __restrict__ x, const float* __restrict__ g,
                          const float* __restrict__ b, float* __restrict__ y) {
    const int row = blockIdx.x;
    const int tid = threadIdx.x;
    const float* xr = x + (size_t)row * EE;
    float v0[4];
    float s1 = 0.f, s2 = 0.f;
#pragma unroll
    for (int i = 0; i < 4; ++i) {
        float t = xr[tid + i * 256];
        v0[i] = t; s1 += t; s2 += t * t;
    }
    __shared__ float r1[256], r2[256];
    r1[tid] = s1; r2[tid] = s2;
    __syncthreads();
    for (int off = 128; off > 0; off >>= 1) {
        if (tid < off) { r1[tid] += r1[tid + off]; r2[tid] += r2[tid + off]; }
        __syncthreads();
    }
    const float mean = r1[0] * (1.0f / EE);
    const float var  = r2[0] * (1.0f / EE) - mean * mean;
    const float rstd = rsqrtf(var + 1e-6f);
    float* yr = y + (size_t)row * EE;
#pragma unroll
    for (int i = 0; i < 4; ++i) {
        int c = tid + i * 256;
        yr[c] = (v0[i] - mean) * rstd * g[c] + b[c];
    }
}

// ---------------- GEMM: out[M,N] = (A[M,K] @ W[N,K]^T + bias (+res) (+ReLU)) * oscale
// 256x128 block tile, BK=32, 256 threads = 8 waves in a 4x2 grid; each wave owns a
// symmetric 64x64 sub-tile (4 A-frags x 4 B-frags -> 16 WMMA per K-step from only
// 8 fragment loads). Global->LDS staging is software-pipelined through registers.
template <bool RELU, bool OUTBF16>
__global__ __launch_bounds__(256) void gemm_kernel(
    const float* __restrict__ A, const float* __restrict__ W,
    const float* __restrict__ bias, const float* res, void* outp,
    int M, int N, int K, float oscale)
{
    __shared__ __align__(16) unsigned As[256 * 18];   // 256 rows x 16 bf16-pairs (+2 pad)
    __shared__ __align__(16) unsigned Ws[128 * 18];
    const int tid  = threadIdx.x;
    const int lane = tid & 31;
    const int wv   = tid >> 5;
    const int half = lane >> 4;
    const int ln   = lane & 15;
    const int wr   = wv >> 1;          // wave row 0..3  (64 rows each)
    const int wc   = wv & 1;           // wave col 0..1  (64 cols each)
    const int n0   = blockIdx.x * 128;
    const int m0   = blockIdx.y * 256;

    const v8f vzero = {};
    v8f acc[4][4];
#pragma unroll
    for (int a2 = 0; a2 < 4; ++a2)
#pragma unroll
        for (int t = 0; t < 4; ++t) acc[a2][t] = vzero;

    float2 ra[16], rw[8];
    auto ldst = [&](int k0) {
#pragma unroll
        for (int i = 0; i < 16; ++i) {          // A tile: 256 rows x 16 pairs
            int idx = tid + i * 256;            // 0..4095
            int r = idx >> 4, cp = idx & 15;
            ra[i] = *(const float2*)(A + (size_t)(m0 + r) * K + k0 + cp * 2);
        }
#pragma unroll
        for (int i = 0; i < 8; ++i) {           // W tile: 128 rows x 16 pairs
            int idx = tid + i * 256;
            int r = idx >> 4, cp = idx & 15;
            rw[i] = *(const float2*)(W + (size_t)(n0 + r) * K + k0 + cp * 2);
        }
    };
    auto stst = [&]() {
#pragma unroll
        for (int i = 0; i < 16; ++i) {
            int idx = tid + i * 256;
            int r = idx >> 4, cp = idx & 15;
            As[r * 18 + cp] = pack2bf(ra[i].x, ra[i].y);
        }
#pragma unroll
        for (int i = 0; i < 8; ++i) {
            int idx = tid + i * 256;
            int r = idx >> 4, cp = idx & 15;
            Ws[r * 18 + cp] = pack2bf(rw[i].x, rw[i].y);
        }
    };

    ldst(0);
    for (int k0 = 0; k0 < K; k0 += 32) {
        __syncthreads();                 // previous compute done with LDS
        stst();
        __syncthreads();
        if (k0 + 32 < K) ldst(k0 + 32);  // overlap next global fetch with WMMA

        Frag16 a[4];
#pragma unroll
        for (int a2 = 0; a2 < 4; ++a2)
#pragma unroll
            for (int v = 0; v < 8; ++v)  // A layout: K-pair = (v&3)*2 + (v>=4?16:0) + half*8
                a[a2].u[v] = As[(wr * 64 + a2 * 16 + ln) * 18 +
                                (v & 3) + ((v >> 2) << 3) + half * 4];
#pragma unroll
        for (int t = 0; t < 4; ++t) {
            Frag16 bf;
#pragma unroll
            for (int v = 0; v < 8; ++v)  // B layout: K-pair = (v&3)*2 + (v>=4?16:0) + half*8
                bf.u[v] = Ws[(wc * 64 + t * 16 + ln) * 18 + (v & 3) + ((v >> 2) << 3) + half * 4];
#pragma unroll
            for (int a2 = 0; a2 < 4; ++a2)
                acc[a2][t] = __builtin_amdgcn_wmma_f32_16x16x32_bf16(
                    false, a[a2].v, false, bf.v, (short)0, acc[a2][t], false, false);
        }
    }

#pragma unroll
    for (int a2 = 0; a2 < 4; ++a2)
#pragma unroll
        for (int t = 0; t < 4; ++t) {
            const int n = n0 + wc * 64 + t * 16 + ln;
            const float bv = bias[n];
#pragma unroll
            for (int r = 0; r < 8; ++r) {
                const int m = m0 + wr * 64 + a2 * 16 + r + half * 8;
                float val = acc[a2][t][r] + bv;
                if (res) val += res[(size_t)m * N + n];
                if (RELU) val = fmaxf(val, 0.f);
                val *= oscale;
                if (OUTBF16) {
                    ((unsigned short*)outp)[(size_t)m * N + n] =
                        __builtin_bit_cast(unsigned short, (__bf16)val);
                } else {
                    ((float*)outp)[(size_t)m * N + n] = val;
                }
            }
        }
}

// ---------------- Flash attention: block = (b, h, 128 q rows), 8 waves x 16 rows ----
// Q/K/V arrive as bf16 (scale folded into Q projection). K chunk staged by the TDM.
__global__ __launch_bounds__(256) void attn_kernel(
    const unsigned short* __restrict__ Qb, const unsigned short* __restrict__ Kb,
    const unsigned short* __restrict__ Vb, float* __restrict__ Z,
    int Lk, int causal)
{
    __shared__ __align__(16) unsigned Ks [64 * 36];   // [key][dh-pair], 144B row (TDM pad)
    __shared__ __align__(16) unsigned VsT[64 * 33];   // [dh][key-pair]
    __shared__ __align__(16) unsigned Ps [8 * 16 * 33];

    const int tid  = threadIdx.x;
    const int lane = tid & 31;
    const int wv   = tid >> 5;
    const int half = lane >> 4;
    const int ln   = lane & 15;
    const int b    = blockIdx.z;
    const int h    = blockIdx.y;
    const int q0   = blockIdx.x * 128;
    const int qrb  = q0 + wv * 16;
    const float NEG = -3.0e38f;

    // Q fragment (16 rows x 64 dh): direct dword loads of bf16 pairs
    Frag16 qf[2];
    {
        const unsigned* qrow = (const unsigned*)(Qb + (size_t)(b * TT + qrb + ln) * EE + h * DHH);
#pragma unroll
        for (int s = 0; s < 2; ++s)
#pragma unroll
            for (int v = 0; v < 8; ++v)
                qf[s].u[v] = qrow[s * 16 + (v & 3) + ((v >> 2) << 3) + half * 4];
    }

    const v8f vzero = {};
    v8f o[4];
#pragma unroll
    for (int t = 0; t < 4; ++t) o[t] = vzero;
    float mrow[8], lrow[8];
#pragma unroll
    for (int r = 0; r < 8; ++r) { mrow[r] = NEG; lrow[r] = 0.f; }

    const int nch = causal ? (q0 + 128) / 64 : (Lk / 64);
    for (int kc = 0; kc < nch; ++kc) {
        __syncthreads();
        const unsigned short* kbase = Kb + (size_t)(b * Lk + kc * 64) * EE + h * DHH;
#if ATHENA_TDM
        if (wv == 0) {
            // 2-D TDM descriptor: 64x64 bf16 tile, row stride 1024 elems,
            // LDS padding 4 dwords per 32 dwords -> 144B LDS row stride.
            const unsigned ldsb = (unsigned)(unsigned long long)(uintptr_t)&Ks[0];
            const unsigned long long ga = (unsigned long long)(uintptr_t)kbase;
            u32x4 g0;
            g0[0] = 1u;                                   // count=1
            g0[1] = ldsb;                                 // lds_addr
            g0[2] = (unsigned)ga;                         // global_addr[31:0]
            g0[3] = (unsigned)((ga >> 32) & 0x01FFFFFFu) | (2u << 30);  // addr hi | type=2
            i32x8 g1;
            g1[0] = (int)((1u << 16) |                    // data_size = 2 bytes
                          (1u << 20) |                    // pad_enable
                          (4u << 22) |                    // pad_interval: 32 dwords
                          (3u << 25));                    // pad_amount : 4 dwords
            g1[1] = 0;                                    // barrier addr / dim0 lo16
            g1[2] = 0x0010;                               // tensor_dim0 = 1<<20 (no OOB)
            g1[3] = (int)((64u << 16) | 0x0010u);         // tile_dim0=64 | tensor_dim1 hi
            g1[4] = 64;                                   // tile_dim1 = 64
            g1[5] = 1024;                                 // tensor_dim0_stride = 1024
            g1[6] = 0; g1[7] = 0;
            i32x4 z4 = {};
#if defined(__clang_major__) && (__clang_major__ >= 23)
            i32x8 z8 = {};
            __builtin_amdgcn_tensor_load_to_lds(g0, g1, z4, z4, z8, 0);
#else
            __builtin_amdgcn_tensor_load_to_lds(g0, g1, z4, z4, 0);
#endif
        }
#else
        // fallback: cooperative dword copy, 64 rows x 32 dwords
#pragma unroll
        for (int i = 0; i < 8; ++i) {
            int idx = tid + i * 256;
            int r = idx >> 5, cp = idx & 31;
            Ks[r * 36 + cp] = ((const unsigned*)(kbase + (size_t)r * EE))[cp];
        }
#endif
        // V chunk transposed (16-bit granularity -> manual): VsT[dh][key-pair]
#pragma unroll
        for (int i = 0; i < 8; ++i) {
            int idx = tid + i * 256;             // 0..2047
            int d = idx & 63, kp = idx >> 6;     // kp in 0..31
            const unsigned short* vp = Vb + (size_t)(b * Lk + kc * 64 + kp * 2) * EE + h * DHH + d;
            VsT[d * 33 + kp] = (unsigned)vp[0] | ((unsigned)vp[EE] << 16);
        }
#if ATHENA_TDM
        if (wv == 0) __builtin_amdgcn_s_wait_tensorcnt(0);
#endif
        __syncthreads();

        // S = Q @ K^T  (16 q rows x 64 keys, 4 C-tiles)
        v8f sacc[4];
#pragma unroll
        for (int t = 0; t < 4; ++t) sacc[t] = vzero;
#pragma unroll
        for (int s = 0; s < 2; ++s)
#pragma unroll
            for (int t = 0; t < 4; ++t) {
                Frag16 bf;
#pragma unroll
                for (int v = 0; v < 8; ++v)
                    bf.u[v] = Ks[(t * 16 + ln) * 36 + s * 16 + (v & 3) + ((v >> 2) << 3) + half * 4];
                sacc[t] = __builtin_amdgcn_wmma_f32_16x16x32_bf16(
                    false, qf[s].v, false, bf.v, (short)0, sacc[t], false, false);
            }

        if (causal) {
#pragma unroll
            for (int t = 0; t < 4; ++t) {
                const int col = kc * 64 + t * 16 + ln;
#pragma unroll
                for (int r = 0; r < 8; ++r)
                    if (col > qrb + r + half * 8) sacc[t][r] = NEG;
            }
        }

        // online softmax (rows live across 16 lanes of each half)
        __bf16* Pw = reinterpret_cast<__bf16*>(&Ps[wv * 16 * 33]);
#pragma unroll
        for (int r = 0; r < 8; ++r) {
            float tm = fmaxf(fmaxf(sacc[0][r], sacc[1][r]), fmaxf(sacc[2][r], sacc[3][r]));
#pragma unroll
            for (int off = 1; off < 16; off <<= 1)
                tm = fmaxf(tm, __shfl_xor(tm, off, 32));
            const float mnew = fmaxf(mrow[r], tm);
            const float sc = __expf(mrow[r] - mnew);
            float ts = 0.f;
#pragma unroll
            for (int t = 0; t < 4; ++t) {
                float p = __expf(sacc[t][r] - mnew);
                ts += p;
                Pw[(r + half * 8) * 66 + t * 16 + ln] = (__bf16)p;
            }
#pragma unroll
            for (int off = 1; off < 16; off <<= 1)
                ts += __shfl_xor(ts, off, 32);
            lrow[r] = lrow[r] * sc + ts;
            mrow[r] = mnew;
#pragma unroll
            for (int t = 0; t < 4; ++t) o[t][r] *= sc;
        }
        __syncthreads();   // make P visible / avoid type-punned reorder

        // O += P @ V   (16 q rows x 64 dh)
#pragma unroll
        for (int s = 0; s < 2; ++s) {
            Frag16 pa;
#pragma unroll
            for (int v = 0; v < 8; ++v)
                pa.u[v] = Ps[wv * 16 * 33 + ln * 33 +
                             s * 16 + (v & 3) + ((v >> 2) << 3) + half * 4];
#pragma unroll
            for (int t = 0; t < 4; ++t) {
                Frag16 bf;
#pragma unroll
                for (int v = 0; v < 8; ++v)
                    bf.u[v] = VsT[(t * 16 + ln) * 33 + s * 16 + (v & 3) + ((v >> 2) << 3) + half * 4];
                o[t] = __builtin_amdgcn_wmma_f32_16x16x32_bf16(
                    false, pa.v, false, bf.v, (short)0, o[t], false, false);
            }
        }
    }

#pragma unroll
    for (int r = 0; r < 8; ++r) {
        const float rinv = 1.0f / lrow[r];
        const int row = qrb + r + half * 8;
        float* zp = Z + (size_t)(b * TT + row) * EE + h * DHH;
#pragma unroll
        for (int t = 0; t < 4; ++t)
            zp[t * 16 + ln] = o[t][r] * rinv;
    }
}

// ---------------- orchestration ----------------
extern "C" void kernel_launch(void* const* d_in, const int* in_sizes, int n_in,
                              void* d_out, int out_size, void* d_ws, size_t ws_size,
                              hipStream_t stream)
{
    (void)in_sizes; (void)n_in; (void)out_size; (void)ws_size;

    const float* tgt   = (const float*)d_in[0];
    const float* z_src = (const float*)d_in[2];
    const float* z_pre = (const float*)d_in[4];
    const int I_SELF = 6, I_SRC = 16, I_PRE = 26;
    const float* ffW1  = (const float*)d_in[36];
    const float* ffb1  = (const float*)d_in[37];
    const float* ffW2  = (const float*)d_in[38];
    const float* ffb2  = (const float*)d_in[39];
    const float* lnffg = (const float*)d_in[40];
    const float* lnffb = (const float*)d_in[41];
    const float* normg = (const float*)d_in[42];
    const float* normb = (const float*)d_in[43];

    // workspace (192MB total):
    //   X(32M f32) | T1(32M f32) | region 128M: {Qb,Kb,Vb bf16 16M each, R3 f32 32M} / H f32
    float* X  = (float*)d_ws;
    float* T1 = X + (size_t)MR * EE;
    unsigned short* Qb = (unsigned short*)(T1 + (size_t)MR * EE);
    unsigned short* Kb = Qb + (size_t)MR * EE;
    unsigned short* Vb = Kb + (size_t)MR * EE;
    float* R3 = (float*)(Vb + (size_t)MR * EE);
    float* H  = (float*)Qb;   // FF hidden reuses the whole region

    hipMemcpyAsync(X, tgt, (size_t)MR * EE * sizeof(float),
                   hipMemcpyDeviceToDevice, stream);

    const dim3 bLN(256), gLN(MR);
    const dim3 bG(256);
    const dim3 gGp(EE / 128, MR / 256);
    const dim3 gGf(FFF / 128, MR / 256);
    const dim3 bA(256), gA(TT / 128, NHH, BB);
    const float SC = 0.125f;   // 1/sqrt(DH)

    struct Mod { const float *Wq,*Wk,*Wv,*Wo,*bq,*bk,*bv,*bo,*lg,*lb; };
    auto getmod = [&](int base, int i) -> Mod {
        Mod m;
        m.Wq = (const float*)d_in[base + 0] + (size_t)i * EE * EE;
        m.Wk = (const float*)d_in[base + 1] + (size_t)i * EE * EE;
        m.Wv = (const float*)d_in[base + 2] + (size_t)i * EE * EE;
        m.Wo = (const float*)d_in[base + 3] + (size_t)i * EE * EE;
        m.bq = (const float*)d_in[base + 4] + (size_t)i * EE;
        m.bk = (const float*)d_in[base + 5] + (size_t)i * EE;
        m.bv = (const float*)d_in[base + 6] + (size_t)i * EE;
        m.bo = (const float*)d_in[base + 7] + (size_t)i * EE;
        m.lg = (const float*)d_in[base + 8] + (size_t)i * EE;
        m.lb = (const float*)d_in[base + 9] + (size_t)i * EE;
        return m;
    };

    auto attn_block = [&](const Mod& m, const float* kvsrc, int causal) {
        ln_kernel<<<gLN, bLN, 0, stream>>>(X, m.lg, m.lb, T1);
        gemm_kernel<false, true ><<<gGp, bG, 0, stream>>>(T1,    m.Wq, m.bq, nullptr, Qb, MR, EE, EE, SC);
        gemm_kernel<false, true ><<<gGp, bG, 0, stream>>>(kvsrc, m.Wk, m.bk, nullptr, Kb, MR, EE, EE, 1.f);
        gemm_kernel<false, true ><<<gGp, bG, 0, stream>>>(kvsrc, m.Wv, m.bv, nullptr, Vb, MR, EE, EE, 1.f);
        attn_kernel<<<gA, bA, 0, stream>>>(Qb, Kb, Vb, R3, TT, causal);
        gemm_kernel<false, false><<<gGp, bG, 0, stream>>>(R3, m.Wo, m.bo, X, X, MR, EE, EE, 1.f);
    };

    for (int i = 0; i < LLL; ++i) {
        attn_block(getmod(I_SELF, i), T1,    1);   // causal self-attn (kv = ln(x))
        attn_block(getmod(I_SRC,  i), z_src, 0);   // cross-attn on z_src
        attn_block(getmod(I_PRE,  i), z_pre, 0);   // cross-attn on z_pre
        // feed-forward
        ln_kernel<<<gLN, bLN, 0, stream>>>(X, lnffg + (size_t)i * EE, lnffb + (size_t)i * EE, T1);
        gemm_kernel<true , false><<<gGf, bG, 0, stream>>>(T1, ffW1 + (size_t)i * FFF * EE,
                                                          ffb1 + (size_t)i * FFF, nullptr, H,
                                                          MR, FFF, EE, 1.f);
        gemm_kernel<false, false><<<gGp, bG, 0, stream>>>(H,  ffW2 + (size_t)i * EE * FFF,
                                                          ffb2 + (size_t)i * EE, X, X,
                                                          MR, EE, FFF, 1.f);
    }
    ln_kernel<<<gLN, bLN, 0, stream>>>(X, normg, normb, (float*)d_out);
}
